// ResidualSimVQ_48919677501490
// MI455X (gfx1250) — matrix operand order repaired and checked by reading, not compiled, not run on hardware
//
#include <hip/hip_runtime.h>
#include <hip/hip_bf16.h>

// ---------------------------------------------------------------------------
// ResidualSimVQ forward for MI455X (gfx1250, wave32, WMMA).
// Shapes: z(16,512,2048)  V(9,512,8)  g(9,512)  b(9,512)  CB(9,1024,8)
// Outputs concatenated in d_out: z_q (16*512*2048) | codes (16*9*2048) | loss (1)
// ---------------------------------------------------------------------------

#define BATCH   16
#define DIM     512
#define TLEN    2048
#define BT      (BATCH * TLEN)     // 32768 rows
#define NCB     9
#define CS      1024
#define CD      8
#define RT_BLK  4                  // row tiles (of 16) per block
#define ROWS_BLK (RT_BLK * 16)     // 64 rows per block

typedef __attribute__((ext_vector_type(16))) __bf16 v16bf;
typedef __attribute__((ext_vector_type(8)))  float  v8f;

union BF16Frag {
    v16bf v;
    uint4 q[2];
    unsigned short s[16];
};

__device__ __forceinline__ unsigned short f2bf(float f) {
    unsigned int u = __float_as_uint(f);
    unsigned int r = u + 0x7FFFu + ((u >> 16) & 1u);   // round-to-nearest-even
    return (unsigned short)(r >> 16);
}

// ---------------------------------------------------------------------------
// (B,D,T) -> (B,T,D) transpose of z into the fp32 residual buffer.
// ---------------------------------------------------------------------------
__global__ __launch_bounds__(256) void k_transpose_in(const float* __restrict__ z,
                                                      float* __restrict__ res) {
    __shared__ float tile[32][33];
    const int b  = blockIdx.z;
    const int t0 = blockIdx.x * 32;
    const int d0 = blockIdx.y * 32;
    const int tx = threadIdx.x, ty = threadIdx.y;          // 32 x 8
    const float* zp = z + (size_t)b * DIM * TLEN;
    for (int i = ty; i < 32; i += 8)
        tile[i][tx] = zp[(size_t)(d0 + i) * TLEN + t0 + tx];
    __syncthreads();
    float* rp = res + (size_t)b * TLEN * DIM;
    for (int i = ty; i < 32; i += 8)
        rp[(size_t)(t0 + i) * DIM + d0 + tx] = tile[tx][i];
}

// (B,T,D) accumulator -> (B,D,T) into d_out.
__global__ __launch_bounds__(256) void k_transpose_out(const float* __restrict__ acc,
                                                       float* __restrict__ out) {
    __shared__ float tile[32][33];
    const int b  = blockIdx.z;
    const int d0 = blockIdx.x * 32;
    const int t0 = blockIdx.y * 32;
    const int tx = threadIdx.x, ty = threadIdx.y;
    const float* ap = acc + (size_t)b * TLEN * DIM;
    for (int i = ty; i < 32; i += 8)
        tile[i][tx] = ap[(size_t)(t0 + i) * DIM + d0 + tx];
    __syncthreads();
    float* op = out + (size_t)b * DIM * TLEN;
    for (int i = ty; i < 32; i += 8)
        op[(size_t)(d0 + i) * TLEN + t0 + tx] = tile[tx][i];
}

__global__ void k_zero_scalar(float* p) { *p = 0.0f; }

__global__ void k_finalize(const float* __restrict__ commit, float* __restrict__ lossOut) {
    *lossOut = *commit;
}

// ---------------------------------------------------------------------------
// Stage prep 1: weight-norm rows  Wn[d][c] = g[d] * V[d][c] / max(||V[d,:]||,1e-12)
// ---------------------------------------------------------------------------
__global__ __launch_bounds__(256) void k_weightnorm(const float* __restrict__ Vi,
                                                    const float* __restrict__ gi,
                                                    float* __restrict__ Wn) {
    const int d = blockIdx.x * blockDim.x + threadIdx.x;
    if (d >= DIM) return;
    float v[CD]; float s = 0.0f;
    #pragma unroll
    for (int c = 0; c < CD; ++c) { v[c] = Vi[d * CD + c]; s += v[c] * v[c]; }
    const float inv = gi[d] / fmaxf(sqrtf(s), 1e-12f);
    #pragma unroll
    for (int c = 0; c < CD; ++c) Wn[d * CD + c] = v[c] * inv;
}

// ---------------------------------------------------------------------------
// Stage prep 2: implicit[s][d] = b[d] + sum_c CB[s][c]*Wn[d][c]
// Writes fp32 row, bf16 row (WMMA B-matrix), and cn[s] = ||implicit_s||^2.
// ---------------------------------------------------------------------------
__global__ __launch_bounds__(256) void k_implicit(const float* __restrict__ CBi,
                                                  const float* __restrict__ Wn,
                                                  const float* __restrict__ bi,
                                                  float* __restrict__ impF,
                                                  unsigned short* __restrict__ impBF,
                                                  float* __restrict__ cn) {
    __shared__ float red[8];
    const int s = blockIdx.x;
    float cb[CD];
    #pragma unroll
    for (int c = 0; c < CD; ++c) cb[c] = CBi[s * CD + c];
    float nrm = 0.0f;
    for (int d = threadIdx.x; d < DIM; d += 256) {
        float val = bi[d];
        #pragma unroll
        for (int c = 0; c < CD; ++c) val += Wn[d * CD + c] * cb[c];
        impF[(size_t)s * DIM + d]  = val;
        impBF[(size_t)s * DIM + d] = f2bf(val);
        nrm += val * val;
    }
    #pragma unroll
    for (int off = 16; off >= 1; off >>= 1) nrm += __shfl_xor(nrm, off, 32);
    const int lane = threadIdx.x & 31, wv = threadIdx.x >> 5;
    if (lane == 0) red[wv] = nrm;
    __syncthreads();
    if (threadIdx.x == 0) {
        float t = 0.0f;
        #pragma unroll
        for (int w = 0; w < 8; ++w) t += red[w];
        cn[s] = t;
    }
}

// ---------------------------------------------------------------------------
// Main WMMA kernel: nearest-code search.
//   scores = res(64xK) @ implicit^T(Kx16 per tile)  via v_wmma_f32_16x16x32_bf16
//   argmin_c ( cn[c] - 2*score )    ( ||res||^2 per row is constant -> dropped )
// Block = 256 threads (8 waves) sharing a 64x512 bf16 res tile in LDS (64 KB,
// 16B-chunk XOR swizzle -> conflict-free A-fragment ds_load_b128).
// kt loop is rolled (unroll 1) and source-unrolled by 2 with two NAMED
// B fragments (b0/b1) -> SSA values stay in registers (no scratch round trip,
// unlike a runtime-indexed bf[2] array), and the compiler can rename across
// the back-edge for true register double-buffering of the global B stream.
// Each B fragment feeds 4 WMMAs -> 4x less L2 traffic for the implicit
// codebook (512 MB/stage instead of 2 GB/stage).
// ---------------------------------------------------------------------------
__global__ __launch_bounds__(256) void k_argmin(const float* __restrict__ res,
                                                const unsigned short* __restrict__ impBF,
                                                const float* __restrict__ cn,
                                                int* __restrict__ idxOut,
                                                float* __restrict__ codesOut,
                                                int stage) {
    __shared__ __align__(16) unsigned char smem[ROWS_BLK * DIM * 2];   // 65536 B
    unsigned short* tile = (unsigned short*)smem;

    const int rowBase = blockIdx.x * ROWS_BLK;
    const int tid = threadIdx.x;

    // stage 64x512 fp32 -> bf16 into LDS with 16B-chunk XOR swizzle:
    // addr(r,k) = r*512 + ((chunk ^ (r&15))*8 | (k&7)),  chunk = k>>3
    for (int e = tid; e < ROWS_BLK * DIM; e += 256) {
        const int r = e >> 9;
        const int k = e & (DIM - 1);
        const int c = k >> 3;
        tile[r * DIM + (((c ^ (r & 15)) << 3) | (k & 7))] =
            f2bf(res[(size_t)(rowBase + r) * DIM + k]);
    }
    __syncthreads();

    const int wave = tid >> 5;
    const int lane = tid & 31;
    const int half = lane >> 4;     // 0: lanes 0-15, 1: lanes 16-31
    const int l15  = lane & 15;

    float best[RT_BLK][8];
    int   bidx[RT_BLK][8];
    #pragma unroll
    for (int rt = 0; rt < RT_BLK; ++rt)
        #pragma unroll
        for (int r = 0; r < 8; ++r) { best[rt][r] = 3.4e38f; bidx[rt][r] = 0; }

    for (int ct = wave; ct < CS / 16; ct += 8) {
        const int c0 = ct * 16;
        v8f acc[RT_BLK];
        #pragma unroll
        for (int rt = 0; rt < RT_BLK; ++rt) acc[rt] = (v8f){};

        // B-matrix base: column N = l15, K-half selected by lane group
        const unsigned short* bptr = impBF + (size_t)(c0 + l15) * DIM + half * 16;

        // two named B fragments (SSA -> registers), loop unrolled by 2 in source
        BF16Frag b0, b1;
        {
            const uint4* bq = reinterpret_cast<const uint4*>(bptr);
            b0.q[0] = bq[0];
            b0.q[1] = bq[1];
        }
        #pragma unroll 1
        for (int kt = 0; kt < DIM / 32; kt += 2) {
            // prefetch B(kt+1) -- always in range since kt is even and < 16
            {
                const uint4* bq = reinterpret_cast<const uint4*>(bptr + (kt + 1) * 32);
                b1.q[0] = bq[0];
                b1.q[1] = bq[1];
            }
            // consume b0 at kb = kt*32
            {
                const int kb  = kt * 32;
                const int ch0 = (kb >> 3) + 2 * half;
                const int ch1 = ch0 + 1;
                #pragma unroll
                for (int rt = 0; rt < RT_BLK; ++rt) {
                    const int row = rt * 16 + l15;        // row & 15 == l15
                    BF16Frag af;
                    af.q[0] = *reinterpret_cast<const uint4*>(&tile[row * DIM + ((ch0 ^ l15) << 3)]);
                    af.q[1] = *reinterpret_cast<const uint4*>(&tile[row * DIM + ((ch1 ^ l15) << 3)]);
                    acc[rt] = __builtin_amdgcn_wmma_f32_16x16x32_bf16(
                                  false, af.v, false, b0.v, (short)0, acc[rt], false, false);
                }
            }
            // prefetch B(kt+2) for the next iteration (uniform branch)
            if (kt + 2 < DIM / 32) {
                const uint4* bq = reinterpret_cast<const uint4*>(bptr + (kt + 2) * 32);
                b0.q[0] = bq[0];
                b0.q[1] = bq[1];
            }
            // consume b1 at kb = (kt+1)*32
            {
                const int kb  = (kt + 1) * 32;
                const int ch0 = (kb >> 3) + 2 * half;
                const int ch1 = ch0 + 1;
                #pragma unroll
                for (int rt = 0; rt < RT_BLK; ++rt) {
                    const int row = rt * 16 + l15;
                    BF16Frag af;
                    af.q[0] = *reinterpret_cast<const uint4*>(&tile[row * DIM + ((ch0 ^ l15) << 3)]);
                    af.q[1] = *reinterpret_cast<const uint4*>(&tile[row * DIM + ((ch1 ^ l15) << 3)]);
                    acc[rt] = __builtin_amdgcn_wmma_f32_16x16x32_bf16(
                                  false, af.v, false, b1.v, (short)0, acc[rt], false, false);
                }
            }
        }
        const float cnv = cn[c0 + l15];
        #pragma unroll
        for (int rt = 0; rt < RT_BLK; ++rt)
            #pragma unroll
            for (int r = 0; r < 8; ++r) {
                const float v = cnv - 2.0f * acc[rt][r];
                if (v < best[rt][r]) { best[rt][r] = v; bidx[rt][r] = c0 + l15; }
            }
    }

    // min-reduce across the 16 column lanes (offsets stay within lane halves)
    #pragma unroll
    for (int off = 8; off >= 1; off >>= 1) {
        #pragma unroll
        for (int rt = 0; rt < RT_BLK; ++rt)
            #pragma unroll
            for (int r = 0; r < 8; ++r) {
                const float ov = __shfl_xor(best[rt][r], off, 32);
                const int   oi = __shfl_xor(bidx[rt][r], off, 32);
                if (ov < best[rt][r] || (ov == best[rt][r] && oi < bidx[rt][r])) {
                    best[rt][r] = ov; bidx[rt][r] = oi;
                }
            }
    }

    // cross-wave reduction: alias the (now dead) tile LDS after a barrier
    __syncthreads();
    float* pv = (float*)smem;                              // [8][64]
    int*   pi = (int*)(smem + 8 * ROWS_BLK * sizeof(float));
    // C/D layout: VGPR r holds row r (lanes 0-15) and row 8+r (lanes 16-31)
    if (l15 == 0) {
        #pragma unroll
        for (int rt = 0; rt < RT_BLK; ++rt)
            #pragma unroll
            for (int r = 0; r < 8; ++r) {
                const int row = rt * 16 + half * 8 + r;
                pv[wave * ROWS_BLK + row] = best[rt][r];
                pi[wave * ROWS_BLK + row] = bidx[rt][r];
            }
    }
    __syncthreads();
    if (tid < ROWS_BLK) {
        float bv = pv[tid]; int bi = pi[tid];
        #pragma unroll
        for (int w = 1; w < 8; ++w) {
            const float v  = pv[w * ROWS_BLK + tid];
            const int   i2 = pi[w * ROWS_BLK + tid];
            if (v < bv || (v == bv && i2 < bi)) { bv = v; bi = i2; }
        }
        const int grow = rowBase + tid;
        idxOut[grow] = bi;
        const int bb = grow >> 11;          // / TLEN
        const int t  = grow & (TLEN - 1);
        codesOut[((size_t)bb * NCB + stage) * TLEN + t] = (float)bi;
    }
}

// ---------------------------------------------------------------------------
// Rotation trick + residual update + commitment loss. One wave per row.
// ---------------------------------------------------------------------------
__global__ __launch_bounds__(256) void k_rotate(float* __restrict__ res,
                                                float* __restrict__ zqacc,
                                                const float* __restrict__ impF,
                                                const float* __restrict__ cn,
                                                const int* __restrict__ idx,
                                                float* __restrict__ commit,
                                                int first) {
    const int wave = threadIdx.x >> 5, lane = threadIdx.x & 31;
    const int row  = blockIdx.x * 8 + wave;
    const int code = idx[row];
    const float* rp = res  + (size_t)row  * DIM;
    const float* qp = impF + (size_t)code * DIM;

    float e[16], q[16];
    float ns2 = 0.0f, dot = 0.0f, dsum = 0.0f;
    #pragma unroll
    for (int j = 0; j < 16; ++j) {
        e[j] = rp[j * 32 + lane];
        q[j] = qp[j * 32 + lane];
        ns2 += e[j] * e[j];
        dot += e[j] * q[j];
        const float df = e[j] - q[j];
        dsum += df * df;
    }
    #pragma unroll
    for (int off = 16; off >= 1; off >>= 1) {
        ns2  += __shfl_xor(ns2,  off, 32);
        dot  += __shfl_xor(dot,  off, 32);
        dsum += __shfl_xor(dsum, off, 32);
    }

    const float nt2 = cn[code];
    const float ns  = sqrtf(ns2), nt = sqrtf(nt2);
    const float nsc = fmaxf(ns, 1e-6f), ntc = fmaxf(nt, 1e-6f);
    // u = e/nsc, qh = zq/ntc, w = u + qh
    const float wn2 = ns2 / (nsc * nsc) + 2.0f * dot / (nsc * ntc) + nt2 / (ntc * ntc);
    const float wnc = fmaxf(sqrtf(wn2), 1e-6f);
    const float ew  = ns2 / nsc + dot / ntc;        // e . w
    const float eu  = ns2 / nsc;                    // e . u
    const float c2  = 2.0f * ew / (wnc * wnc);
    const float c3  = 2.0f * eu / ntc;
    const float scale = nt / nsc;

    float* ap = zqacc + (size_t)row * DIM;
    float* wp = res   + (size_t)row * DIM;
    #pragma unroll
    for (int j = 0; j < 16; ++j) {
        const float wj  = e[j] / nsc + q[j] / ntc;
        const float rot = e[j] - c2 * wj + c3 * q[j];
        const float zqi = rot * scale;
        const float prev = first ? 0.0f : ap[j * 32 + lane];
        ap[j * 32 + lane] = prev + zqi;
        wp[j * 32 + lane] = e[j] - zqi;
    }
    if (lane == 0)
        atomicAdd(commit, 1.25f * dsum / ((float)BT * (float)DIM));
}

// ---------------------------------------------------------------------------
extern "C" void kernel_launch(void* const* d_in, const int* in_sizes, int n_in,
                              void* d_out, int out_size, void* d_ws, size_t ws_size,
                              hipStream_t stream) {
    const float* z  = (const float*)d_in[0];
    const float* V  = (const float*)d_in[1];
    const float* g  = (const float*)d_in[2];
    const float* b  = (const float*)d_in[3];
    const float* CB = (const float*)d_in[4];

    float* out = (float*)d_out;
    float* outCodes = out + (size_t)BATCH * DIM * TLEN;            // 16777216
    float* outLoss  = outCodes + (size_t)BATCH * NCB * TLEN;       // +294912

    // workspace carve-up (all 16B aligned)
    float*          res    = (float*)d_ws;                         // BT*DIM
    float*          zqacc  = res   + (size_t)BT * DIM;             // BT*DIM
    float*          impF   = zqacc + (size_t)BT * DIM;             // CS*DIM
    unsigned short* impBF  = (unsigned short*)(impF + (size_t)CS * DIM);  // CS*DIM bf16
    float*          cn     = (float*)(impBF + (size_t)CS * DIM);   // CS
    float*          Wn     = cn + CS;                              // DIM*CD
    int*            idxBuf = (int*)(Wn + DIM * CD);                // BT
    float*          commit = (float*)(idxBuf + BT);                // 1

    k_transpose_in<<<dim3(TLEN / 32, DIM / 32, BATCH), dim3(32, 8), 0, stream>>>(z, res);
    k_zero_scalar<<<1, 1, 0, stream>>>(commit);

    for (int i = 0; i < NCB; ++i) {
        k_weightnorm<<<2, 256, 0, stream>>>(V + (size_t)i * DIM * CD, g + (size_t)i * DIM, Wn);
        k_implicit<<<CS, 256, 0, stream>>>(CB + (size_t)i * CS * CD, Wn, b + (size_t)i * DIM,
                                           impF, impBF, cn);
        k_argmin<<<BT / ROWS_BLK, 256, 0, stream>>>(res, impBF, cn, idxBuf, outCodes, i);
        k_rotate<<<BT / 8, 256, 0, stream>>>(res, zqacc, impF, cn, idxBuf, commit, i == 0);
    }

    k_transpose_out<<<dim3(DIM / 32, TLEN / 32, BATCH), dim3(32, 8), 0, stream>>>(zqacc, out);
    k_finalize<<<1, 1, 0, stream>>>(commit, outLoss);
}
